// ChebConv_14766097564103
// MI455X (gfx1250) — matrix operand order repaired and checked by reading, not compiled, not run on hardware
//
#include <hip/hip_runtime.h>
#include <hip/hip_bf16.h>
#include <math.h>

typedef __attribute__((ext_vector_type(16))) _Float16 v16h;
typedef __attribute__((ext_vector_type(8)))  _Float16 v8h;
typedef __attribute__((ext_vector_type(8)))  float    v8f;
typedef int v4i __attribute__((vector_size(16)));

typedef __attribute__((address_space(1))) v4i* gptr_v4i;
typedef __attribute__((address_space(3))) v4i* lptr_v4i;

#define WMMA16(a, b, c) \
  __builtin_amdgcn_wmma_f32_16x16x32_f16(false, (a), false, (b), (short)0, (c), false, false)

#define CAT16(lo, hi) \
  __builtin_shufflevector((lo), (hi), 0,1,2,3,4,5,6,7,8,9,10,11,12,13,14,15)

#if __has_builtin(__builtin_amdgcn_global_load_async_to_lds_b128)
#define HAS_ASYNC_LDS 1
#else
#define HAS_ASYNC_LDS 0
#endif

#define WAIT_ASYNC() asm volatile("s_wait_asynccnt 0x0" ::: "memory")

static constexpr int Bn   = 16;
static constexpr int Nn   = 2048;
static constexpr int Fn   = 64;
static constexpr int OutD = 128;

// ---------------------------------------------------------------------------
// 1) invd[b,i] = 1 / (eps + sqrt(rowsum(A with zero diagonal)))
// ---------------------------------------------------------------------------
__global__ __launch_bounds__(256) void rowsum_kernel(const float* __restrict__ A,
                                                     float* __restrict__ invd) {
  const int gw   = (blockIdx.x * blockDim.x + threadIdx.x) >> 5;  // row id in [0, B*N)
  const int lane = threadIdx.x & 31;
  const float4* row4 = (const float4*)(A + (size_t)gw * Nn);
  float s = 0.f;
#pragma unroll
  for (int it = 0; it < Nn / 128; ++it) {
    float4 f = row4[lane + 32 * it];
    s += f.x + f.y + f.z + f.w;
  }
#pragma unroll
  for (int off = 16; off > 0; off >>= 1) s += __shfl_xor(s, off, 32);
  const int i = gw & (Nn - 1);
  s -= ((const float*)row4)[i];
  if (lane == 0) invd[gw] = 1.0f / (1e-6f + sqrtf(s));
}

// ---------------------------------------------------------------------------
// 2a) Pass 1: T1 = L @ X with L formed on the fly from fp32 A; ALSO writes the
//     normalized L out as f16 (128 MB total -> L2-resident for passes 2/3).
// ---------------------------------------------------------------------------
__global__ __launch_bounds__(256) void cheb_gemm_first_kernel(
    const float* __restrict__ A, const float* __restrict__ invd,
    const float* __restrict__ X, float* __restrict__ T1,
    _Float16* __restrict__ Lg) {
  __shared__ _Float16 Lt[128 * 40];
  __shared__ _Float16 Xt[64 * 40];

  const int b       = blockIdx.y;
  const int rowbase = blockIdx.x * 128;
  const float* Ab   = A    + (size_t)b * Nn * Nn;
  const float* db   = invd + (size_t)b * Nn;
  const float* Xb   = X    + (size_t)b * Nn * Fn;
  float*       T1b  = T1   + (size_t)b * Nn * Fn;
  _Float16*    Lgb  = Lg   + (size_t)b * Nn * Nn;

  const int t = threadIdx.x;
  const int w = t >> 5, lane = t & 31, h = lane >> 4, mrow = lane & 15;
  const int sr = t >> 1, sk = (t & 1) * 16;
  const int gr = rowbase + sr;
  const float drow = db[gr];
  const int xk = t >> 3, xc = (t & 7) * 8;

  v8f acc[4] = {};

  for (int kb = 0; kb < Nn; kb += 32) {
    const float4* ap = (const float4*)(Ab + (size_t)gr * Nn + kb + sk);
    float4 a0 = ap[0], a1 = ap[1], a2 = ap[2], a3 = ap[3];
    const float4* dp = (const float4*)(db + kb + sk);
    float4 d0 = dp[0], d1 = dp[1], d2 = dp[2], d3 = dp[3];
    const float4* xp = (const float4*)(Xb + (size_t)(kb + xk) * Fn + xc);
    float4 x0 = xp[0], x1 = xp[1];
    __builtin_prefetch(Ab + (size_t)gr * Nn + kb + 32, 0, 1);

    __syncthreads();

    {   // normalized L tile -> LDS (f16) and -> global f16 L
      float av[16]; ((float4*)av)[0] = a0; ((float4*)av)[1] = a1;
                    ((float4*)av)[2] = a2; ((float4*)av)[3] = a3;
      float dv[16]; ((float4*)dv)[0] = d0; ((float4*)dv)[1] = d1;
                    ((float4*)dv)[2] = d2; ((float4*)dv)[3] = d3;
      union { _Float16 h[16]; v8h v[2]; } u;
#pragma unroll
      for (int j = 0; j < 16; ++j) {
        float v = av[j] * drow * dv[j];
        if (gr == kb + sk + j) v = 0.f;
        u.h[j] = (_Float16)v;
      }
      _Float16* lrow = Lt + sr * 40 + sk;
#pragma unroll
      for (int j = 0; j < 16; ++j) lrow[j] = u.h[j];
      _Float16* gdst = Lgb + (size_t)gr * Nn + kb + sk;
      *(v8h*)gdst       = u.v[0];
      *(v8h*)(gdst + 8) = u.v[1];
    }
    {   // X tile transposed -> LDS (f16)
      float xv[8]; ((float4*)xv)[0] = x0; ((float4*)xv)[1] = x1;
#pragma unroll
      for (int j = 0; j < 8; ++j) Xt[(xc + j) * 40 + xk] = (_Float16)xv[j];
    }
    __syncthreads();

    const _Float16* lr = Lt + (w * 16 + mrow) * 40;
    v8h alo = *(const v8h*)(lr + h * 8);
    v8h ahi = *(const v8h*)(lr + 16 + h * 8);
    v16h afr = CAT16(alo, ahi);
#pragma unroll
    for (int c = 0; c < 4; ++c) {
      const _Float16* xr = Xt + (c * 16 + mrow) * 40 + h * 8;
      v16h bfr = CAT16(*(const v8h*)xr, *(const v8h*)(xr + 16));
      acc[c] = WMMA16(afr, bfr, acc[c]);
    }
  }

#pragma unroll
  for (int c = 0; c < 4; ++c)
#pragma unroll
    for (int v = 0; v < 8; ++v) {
      const int grow = rowbase + w * 16 + v + 8 * h;
      const int col  = c * 16 + mrow;
      T1b[(size_t)grow * Fn + col] = acc[c][v];
    }
}

// ---------------------------------------------------------------------------
// 2b) Passes 2/3: Tout = 2 * (Lf16 @ Tin) - Tprev.  L tile is DMA'd into LDS
//     with GLOBAL_LOAD_ASYNC_TO_LDS_B128 (ASYNCcnt path).
// ---------------------------------------------------------------------------
__global__ __launch_bounds__(256) void cheb_gemm_f16_kernel(
    const _Float16* __restrict__ Lg, const float* __restrict__ Tin,
    const float* __restrict__ Tprev, float* __restrict__ Tout) {
  __shared__ _Float16 Lt[128 * 40];
  __shared__ _Float16 Xt[64 * 40];

  const int b       = blockIdx.y;
  const int rowbase = blockIdx.x * 128;
  const _Float16* Lgb = Lg + (size_t)b * Nn * Nn;
  const float* Tinb = Tin  + (size_t)b * Nn * Fn;
  const float* Tpb  = Tprev + (size_t)b * Nn * Fn;
  float*       Tob  = Tout + (size_t)b * Nn * Fn;

  const int t = threadIdx.x;
  const int w = t >> 5, lane = t & 31, h = lane >> 4, mrow = lane & 15;
  const int sr = t >> 1, sk = (t & 1) * 16;   // L tile: row, 16-half chunk base
  const int gr = rowbase + sr;
  const int xk = t >> 3, xc = (t & 7) * 8;

  v8f acc[4] = {};

  for (int kb = 0; kb < Nn; kb += 32) {
    const float4* xp = (const float4*)(Tinb + (size_t)(kb + xk) * Fn + xc);
    float4 x0 = xp[0], x1 = xp[1];
    const _Float16* gsrc = Lgb + (size_t)gr * Nn + kb + sk;
    _Float16* ldst = Lt + sr * 40 + sk;
#if !HAS_ASYNC_LDS
    v8h l0 = *(const v8h*)gsrc;
    v8h l1 = *(const v8h*)(gsrc + 8);
#endif
    __builtin_prefetch(gsrc + 32, 0, 1);

    __syncthreads();   // previous tile fully consumed before any LDS write

#if HAS_ASYNC_LDS
    __builtin_amdgcn_global_load_async_to_lds_b128(
        (gptr_v4i)gsrc, (lptr_v4i)ldst, 0, 0);
    __builtin_amdgcn_global_load_async_to_lds_b128(
        (gptr_v4i)(gsrc + 8), (lptr_v4i)(ldst + 8), 0, 0);
#else
    *(v8h*)ldst       = l0;
    *(v8h*)(ldst + 8) = l1;
#endif
    {
      float xv[8]; ((float4*)xv)[0] = x0; ((float4*)xv)[1] = x1;
#pragma unroll
      for (int j = 0; j < 8; ++j) Xt[(xc + j) * 40 + xk] = (_Float16)xv[j];
    }
    WAIT_ASYNC();      // this wave's DMA into LDS has landed
    __syncthreads();

    const _Float16* lr = Lt + (w * 16 + mrow) * 40;
    v8h alo = *(const v8h*)(lr + h * 8);
    v8h ahi = *(const v8h*)(lr + 16 + h * 8);
    v16h afr = CAT16(alo, ahi);
#pragma unroll
    for (int c = 0; c < 4; ++c) {
      const _Float16* xr = Xt + (c * 16 + mrow) * 40 + h * 8;
      v16h bfr = CAT16(*(const v8h*)xr, *(const v8h*)(xr + 16));
      acc[c] = WMMA16(afr, bfr, acc[c]);
    }
  }

#pragma unroll
  for (int c = 0; c < 4; ++c)
#pragma unroll
    for (int v = 0; v < 8; ++v) {
      const int grow = rowbase + w * 16 + v + 8 * h;
      const int col  = c * 16 + mrow;
      const size_t off = (size_t)grow * Fn + col;
      Tob[off] = 2.0f * acc[c][v] - Tpb[off];
    }
}

// ---------------------------------------------------------------------------
// 3) transpose kernel weights [256][128] f32 -> [128][256] f16
// ---------------------------------------------------------------------------
__global__ __launch_bounds__(256) void kt_kernel(const float* __restrict__ kern,
                                                 _Float16* __restrict__ kt) {
  const int idx = blockIdx.x * 256 + threadIdx.x;
  const int k = idx >> 7, n = idx & 127;
  kt[n * 256 + k] = (_Float16)kern[idx];
}

// ---------------------------------------------------------------------------
// 4) out = relu(concat(X,T1,T2,T3) @ kernel + bias), concat never materialized
// ---------------------------------------------------------------------------
__global__ __launch_bounds__(256) void out_gemm_kernel(
    const float* __restrict__ X,  const float* __restrict__ T1,
    const float* __restrict__ T2, const float* __restrict__ T3,
    const _Float16* __restrict__ kt, const float* __restrict__ bias,
    float* __restrict__ out) {
  const int t = threadIdx.x;
  const int w = t >> 5, lane = t & 31, h = lane >> 4, mrow = lane & 15;
  const int rbase = blockIdx.x * 128 + w * 16;

  v8f acc[8] = {};
#pragma unroll
  for (int s = 0; s < 8; ++s) {
    const float* src = (s < 2) ? X : (s < 4) ? T1 : (s < 6) ? T2 : T3;
    const int fb = (s & 1) * 32;
    const float* ap = src + (size_t)(rbase + mrow) * Fn + fb + h * 8;
    float4 f0 = ((const float4*)ap)[0];
    float4 f1 = ((const float4*)ap)[1];
    float4 f2 = ((const float4*)(ap + 16))[0];
    float4 f3 = ((const float4*)(ap + 16))[1];
    float av[16]; ((float4*)av)[0] = f0; ((float4*)av)[1] = f1;
                  ((float4*)av)[2] = f2; ((float4*)av)[3] = f3;
    v16h afr;
#pragma unroll
    for (int j = 0; j < 16; ++j) afr[j] = (_Float16)av[j];
#pragma unroll
    for (int c = 0; c < 8; ++c) {
      const _Float16* bp = kt + (size_t)(c * 16 + mrow) * 256 + s * 32 + h * 8;
      v16h bfr = CAT16(*(const v8h*)bp, *(const v8h*)(bp + 16));
      acc[c] = WMMA16(afr, bfr, acc[c]);
    }
  }

#pragma unroll
  for (int c = 0; c < 8; ++c)
#pragma unroll
    for (int v = 0; v < 8; ++v) {
      const int grow = rbase + v + 8 * h;
      const int col  = c * 16 + mrow;
      float val = acc[c][v] + bias[col];
      out[(size_t)grow * OutD + col] = val > 0.f ? val : 0.f;
    }
}

// ---------------------------------------------------------------------------
extern "C" void kernel_launch(void* const* d_in, const int* in_sizes, int n_in,
                              void* d_out, int out_size, void* d_ws, size_t ws_size,
                              hipStream_t stream) {
  const float* X    = (const float*)d_in[0];   // [16,2048,64]
  const float* A    = (const float*)d_in[1];   // [16,2048,2048]
  const float* kern = (const float*)d_in[2];   // [256,128]
  const float* bias = (const float*)d_in[3];   // [128]
  float* out = (float*)d_out;                  // [16,2048,128]

  char* ws = (char*)d_ws;
  float*    invd = (float*)ws;                                   // 128 KB
  float*    T1   = (float*)(ws + (size_t)(1u << 20));            // 8 MB
  float*    T2   = (float*)(ws + (size_t)(9u << 20));            // 8 MB
  float*    T3   = (float*)(ws + (size_t)(17u << 20));           // 8 MB
  _Float16* kt   = (_Float16*)(ws + (size_t)(25u << 20));        // 64 KB
  _Float16* Lg   = (_Float16*)(ws + (size_t)(26u << 20));        // 128 MB f16 L

  rowsum_kernel<<<dim3(Bn * Nn / 8), 256, 0, stream>>>(A, invd);
  kt_kernel<<<dim3(128), 256, 0, stream>>>(kern, kt);
  // T1 = L @ X  (also materializes f16 L)
  cheb_gemm_first_kernel<<<dim3(Nn / 128, Bn), 256, 0, stream>>>(A, invd, X, T1, Lg);
  // T2 = 2 L @ T1 - X
  cheb_gemm_f16_kernel<<<dim3(Nn / 128, Bn), 256, 0, stream>>>(Lg, T1, X, T2);
  // T3 = 2 L @ T2 - T1
  cheb_gemm_f16_kernel<<<dim3(Nn / 128, Bn), 256, 0, stream>>>(Lg, T2, T1, T3);
  out_gemm_kernel<<<dim3(Bn * Nn / 128), 256, 0, stream>>>(X, T1, T2, T3, kt, bias, out);
}
